// BasePixelDecoder_74277164417674
// MI455X (gfx1250) — matrix operand order repaired and checked by reading, not compiled, not run on hardware
//
#include <hip/hip_runtime.h>
#include <hip/hip_bf16.h>

typedef _Float16 v16h __attribute__((ext_vector_type(16)));
typedef float    v8f  __attribute__((ext_vector_type(8)));

#define EPSV 1e-5f

#define WMMA_F16(A, B, C) \
    __builtin_amdgcn_wmma_f32_16x16x32_f16(false, (A), false, (B), (short)0, (C), false, false)

// ---------------------------------------------------------------------------
// WMMA GEMM:  C[m,n] = sum_k A[m,k] * B[k,n]  (+ bias[m])
//   A : [M,K] f32 row-major weights (shared across batch)
//   B : [K,N] f32 row-major activations, batch stride bBS
//   C : [M,N] f32 row-major, batch stride cBS
// Block tile 64(M) x 128(N), k-step 32. 8 waves in a 2x4 grid, each wave owns a
// 32x32 output patch = 2x2 WMMA tiles (4 v_wmma per k-step, fragments reused).
// Register-pipelined staging: global loads for slab t+1 issue before the WMMA
// work of slab t, f32->f16 conversion happens on the LDS store.
// ---------------------------------------------------------------------------
__global__ __launch_bounds__(256)
void gemm_wmma_kernel(const float* __restrict__ A, const float* __restrict__ B,
                      const float* __restrict__ bias, float* __restrict__ C,
                      int M, int K, int N, long long bBS, long long cBS)
{
    __shared__ _Float16 As[64][40];    // [m][k]   (row stride 80B, 16B aligned)
    __shared__ _Float16 BsT[128][40];  // [n][k]

    const int t    = threadIdx.x;
    const int lane = t & 31;
    const int wave = t >> 5;
    const int wm   = wave & 1;         // M 32-slab of wave
    const int wn   = wave >> 1;        // N 32-slab of wave
    const int bm   = blockIdx.y * 64;
    const int bn   = blockIdx.x * 128;
    const float* Bb = B + (long long)blockIdx.z * bBS;
    float*       Cb = C + (long long)blockIdx.z * cBS;
    const int mloc = lane & 15;
    const int half = lane >> 4;

    // staging decomposition
    const int ar  = t >> 2;            // 0..63  A row
    const int ak  = (t & 3) * 8;       // 0,8,16,24
    const int bk  = t >> 3;            // 0..31  B k-row
    const int bnn = (t & 7) * 16;      // 0..112

    float4 ra0, ra1, rb0, rb1, rb2, rb3;
    const bool fullM = (bm + 64 <= M);

    auto loadTile = [&](int k0) {
        if (fullM && (k0 + 32 <= K)) {
            const float* ap = A + (size_t)(bm + ar) * K + k0 + ak;
            ra0 = *(const float4*)(ap);
            ra1 = *(const float4*)(ap + 4);
            const float* bp = Bb + (size_t)(k0 + bk) * N + bn + bnn;
            rb0 = *(const float4*)(bp);
            rb1 = *(const float4*)(bp + 4);
            rb2 = *(const float4*)(bp + 8);
            rb3 = *(const float4*)(bp + 12);
        } else {
            float4 z = make_float4(0.f, 0.f, 0.f, 0.f);
            ra0 = ra1 = rb0 = rb1 = rb2 = rb3 = z;
            if (bm + ar < M) {          // K is always a multiple of 16
                const float* ap = A + (size_t)(bm + ar) * K;
                if (k0 + ak + 4 <= K) ra0 = *(const float4*)(ap + k0 + ak);
                if (k0 + ak + 8 <= K) ra1 = *(const float4*)(ap + k0 + ak + 4);
            }
            if (k0 + bk < K) {
                const float* bp = Bb + (size_t)(k0 + bk) * N + bn + bnn;
                rb0 = *(const float4*)(bp);
                rb1 = *(const float4*)(bp + 4);
                rb2 = *(const float4*)(bp + 8);
                rb3 = *(const float4*)(bp + 12);
            }
        }
    };

    auto storeTile = [&]() {
        As[ar][ak + 0] = (_Float16)ra0.x; As[ar][ak + 1] = (_Float16)ra0.y;
        As[ar][ak + 2] = (_Float16)ra0.z; As[ar][ak + 3] = (_Float16)ra0.w;
        As[ar][ak + 4] = (_Float16)ra1.x; As[ar][ak + 5] = (_Float16)ra1.y;
        As[ar][ak + 6] = (_Float16)ra1.z; As[ar][ak + 7] = (_Float16)ra1.w;
        BsT[bnn +  0][bk] = (_Float16)rb0.x; BsT[bnn +  1][bk] = (_Float16)rb0.y;
        BsT[bnn +  2][bk] = (_Float16)rb0.z; BsT[bnn +  3][bk] = (_Float16)rb0.w;
        BsT[bnn +  4][bk] = (_Float16)rb1.x; BsT[bnn +  5][bk] = (_Float16)rb1.y;
        BsT[bnn +  6][bk] = (_Float16)rb1.z; BsT[bnn +  7][bk] = (_Float16)rb1.w;
        BsT[bnn +  8][bk] = (_Float16)rb2.x; BsT[bnn +  9][bk] = (_Float16)rb2.y;
        BsT[bnn + 10][bk] = (_Float16)rb2.z; BsT[bnn + 11][bk] = (_Float16)rb2.w;
        BsT[bnn + 12][bk] = (_Float16)rb3.x; BsT[bnn + 13][bk] = (_Float16)rb3.y;
        BsT[bnn + 14][bk] = (_Float16)rb3.z; BsT[bnn + 15][bk] = (_Float16)rb3.w;
    };

    v8f acc00 = {}, acc01 = {}, acc10 = {}, acc11 = {};

    loadTile(0);
    for (int k0 = 0; k0 < K; k0 += 32) {
        storeTile();                 // waits on in-flight loads, converts, stores
        __syncthreads();
        if (k0 + 32 < K) loadTile(k0 + 32);      // overlap with WMMA below
        if (t == 0 && k0 + 64 < K) {             // warm L2 for slab t+2
            __builtin_prefetch(A + (size_t)bm * K + k0 + 64, 0, 1);
            __builtin_prefetch(Bb + (size_t)(k0 + 64) * N + bn, 0, 1);
        }

        v16h af0, af1, bf0, bf1;
        #pragma unroll
        for (int v = 0; v < 8; ++v) {
            int ka = ((v & 4) ? 16 : 0) + half * 8 + (v & 3) * 2;
            af0[2*v+0] = As[wm*32 + mloc][ka];       af0[2*v+1] = As[wm*32 + mloc][ka+1];
            af1[2*v+0] = As[wm*32 + 16 + mloc][ka];  af1[2*v+1] = As[wm*32 + 16 + mloc][ka+1];
            int kb = ((v & 4) ? 16 : 0) + half * 8 + (v & 3) * 2;
            bf0[2*v+0] = BsT[wn*32 + mloc][kb];      bf0[2*v+1] = BsT[wn*32 + mloc][kb+1];
            bf1[2*v+0] = BsT[wn*32 + 16 + mloc][kb]; bf1[2*v+1] = BsT[wn*32 + 16 + mloc][kb+1];
        }
        acc00 = WMMA_F16(af0, bf0, acc00);
        acc01 = WMMA_F16(af0, bf1, acc01);
        acc10 = WMMA_F16(af1, bf0, acc10);
        acc11 = WMMA_F16(af1, bf1, acc11);
        __syncthreads();
    }

    #pragma unroll
    for (int v = 0; v < 8; ++v) {
        int r0 = bm + wm * 32 + v + 8 * half;
        int r1 = r0 + 16;
        int c0 = bn + wn * 32 + mloc;
        int c1 = c0 + 16;
        if (r0 < M) {
            float bb = bias ? bias[r0] : 0.f;
            Cb[(size_t)r0 * N + c0] = acc00[v] + bb;
            Cb[(size_t)r0 * N + c1] = acc01[v] + bb;
        }
        if (r1 < M) {
            float bb = bias ? bias[r1] : 0.f;
            Cb[(size_t)r1 * N + c0] = acc10[v] + bb;
            Cb[(size_t)r1 * N + c1] = acc11[v] + bb;
        }
    }
}

// ---------------------------------------------------------------------------
// GroupNorm(+ReLU), three-stage for parallelism: partial stats, finalize, apply.
// stats[] : [64 groups * 8 splits][2], mi[] : [64][2] = (mean, inv)
// ---------------------------------------------------------------------------
__global__ __launch_bounds__(256)
void gn_stat_kernel(const float* __restrict__ x, float* __restrict__ stats,
                    int C, int HW, long long xBS)
{
    int bg = blockIdx.x;                  // batch*32 + group
    int sp = blockIdx.y;                  // 0..7
    int cpg = C / 32;
    int n = cpg * HW;
    const float* xb = x + (long long)(bg >> 5) * xBS + (size_t)(bg & 31) * n;
    int chunk = (n + 7) / 8;
    int lo = sp * chunk, hi = min(n, lo + chunk);
    __shared__ float s1[256], s2[256];
    float a1 = 0.f, a2 = 0.f;
    for (int i = lo + threadIdx.x; i < hi; i += 256) { float v = xb[i]; a1 += v; a2 += v * v; }
    s1[threadIdx.x] = a1; s2[threadIdx.x] = a2;
    __syncthreads();
    for (int s = 128; s > 0; s >>= 1) {
        if (threadIdx.x < s) { s1[threadIdx.x] += s1[threadIdx.x + s]; s2[threadIdx.x] += s2[threadIdx.x + s]; }
        __syncthreads();
    }
    if (threadIdx.x == 0) { stats[(bg * 8 + sp) * 2] = s1[0]; stats[(bg * 8 + sp) * 2 + 1] = s2[0]; }
}

__global__ void gn_fin_kernel(const float* __restrict__ stats, float* __restrict__ mi,
                              int C, int HW)
{
    int bg = threadIdx.x;
    if (bg >= 64) return;
    float s = 0.f, ss = 0.f;
    for (int p = 0; p < 8; ++p) { s += stats[(bg * 8 + p) * 2]; ss += stats[(bg * 8 + p) * 2 + 1]; }
    float n = (float)((C / 32) * HW);
    float mean = s / n;
    float var  = ss / n - mean * mean;
    mi[bg * 2] = mean;
    mi[bg * 2 + 1] = rsqrtf(var + EPSV);
}

__global__ void gn_apply_kernel(const float* __restrict__ x, float* __restrict__ y,
                                const float* __restrict__ g, const float* __restrict__ b,
                                const float* __restrict__ mi, int C, int HW,
                                long long xBS, long long yBS)
{
    size_t i = (size_t)blockIdx.x * blockDim.x + threadIdx.x;
    size_t per = (size_t)C * HW;
    if (i >= 2 * per) return;
    int batch = (int)(i / per);
    size_t rem = i - (size_t)batch * per;
    int ch = (int)(rem / HW);
    int bg = batch * 32 + ch / (C / 32);
    float v = (x[(long long)batch * xBS + rem] - mi[bg * 2]) * mi[bg * 2 + 1] * g[ch] + b[ch];
    y[(long long)batch * yBS + rem] = fmaxf(v, 0.f);
}

// ---------------------------------------------------------------------------
// BatchNorm (inference) + ReLU, elementwise. grid.z = batch.
// ---------------------------------------------------------------------------
__global__ void bn_relu_kernel(const float* __restrict__ x, float* __restrict__ y,
                               const float* __restrict__ g, const float* __restrict__ b,
                               const float* __restrict__ m, const float* __restrict__ v,
                               int C, int HW, long long xBS, long long yBS)
{
    size_t i = (size_t)blockIdx.x * blockDim.x + threadIdx.x;
    if (i >= (size_t)C * HW) return;
    int ch = (int)(i / HW);
    float sc = g[ch] * rsqrtf(v[ch] + EPSV);
    float val = (x[(long long)blockIdx.z * xBS + i] - m[ch]) * sc + b[ch];
    y[(long long)blockIdx.z * yBS + i] = fmaxf(val, 0.f);
}

// ---------------------------------------------------------------------------
// Build fused 1536-channel input: [c4 up x4 | c3 up x2 | c2] at 64x64.
// ---------------------------------------------------------------------------
__global__ void fin_build_kernel(const float* __restrict__ h4, const float* __restrict__ h3,
                                 const float* __restrict__ h2, float* __restrict__ fin)
{
    size_t id = (size_t)blockIdx.x * blockDim.x + threadIdx.x;
    if (id >= (size_t)2 * 1536 * 4096) return;
    int w = (int)(id & 63), h = (int)((id >> 6) & 63);
    int rest = (int)(id >> 12);
    int c = rest % 1536, b = rest / 1536;
    float v;
    if (c < 512)       v = h4[(((size_t)b * 512 + c) * 16 + (h >> 2)) * 16 + (w >> 2)];
    else if (c < 1024) v = h3[(((size_t)b * 512 + (c - 512)) * 32 + (h >> 1)) * 32 + (w >> 1)];
    else               v = h2[(((size_t)b * 512 + (c - 1024)) * 64 + h) * 64 + w];
    fin[id] = v;
}

// q buffer: qbuf[c][gtok], gtok = ((b*8+nh)*8+nw)*64 + i*8 + j
__global__ void qbuf_build_kernel(const float* __restrict__ cbuf, float* __restrict__ qbuf)
{
    size_t id = (size_t)blockIdx.x * blockDim.x + threadIdx.x;
    if (id >= (size_t)2 * 512 * 4096) return;
    int w = (int)(id & 63), h = (int)((id >> 6) & 63);
    int c = (int)((id >> 12) & 511), b = (int)(id >> 21);
    int gtok = (((b * 8 + (h >> 3)) * 8 + (w >> 3)) << 6) | ((h & 7) * 8 + (w & 7));
    qbuf[(size_t)c * 8192 + gtok] = cbuf[id];
}

// ---------------------------------------------------------------------------
// Pool path
// ---------------------------------------------------------------------------
__global__ __launch_bounds__(256)
void pool_mean_kernel(const float* __restrict__ x, float* __restrict__ pooled)
{
    __shared__ float s[256];
    const float* xb = x + (size_t)blockIdx.x * 4096;
    float a = 0.f;
    for (int i = threadIdx.x; i < 4096; i += 256) a += xb[i];
    s[threadIdx.x] = a; __syncthreads();
    for (int st = 128; st > 0; st >>= 1) {
        if (threadIdx.x < st) s[threadIdx.x] += s[threadIdx.x + st];
        __syncthreads();
    }
    if (threadIdx.x == 0) pooled[blockIdx.x] = s[0] * (1.f / 4096.f);
}

__global__ void pool_fc_kernel(const float* __restrict__ w, const float* __restrict__ pooled,
                               float* __restrict__ out)
{
    int id = blockIdx.x * blockDim.x + threadIdx.x;
    if (id >= 1024) return;
    int b = id >> 9, o = id & 511;
    float acc = 0.f;
    for (int c = 0; c < 512; ++c) acc += w[o * 512 + c] * pooled[b * 512 + c];
    out[id] = acc;
}

__global__ void pool_broadcast_kernel(const float* __restrict__ pn, float* __restrict__ outs)
{
    size_t id = (size_t)blockIdx.x * blockDim.x + threadIdx.x;
    if (id >= (size_t)2 * 512 * 4096) return;
    int p = (int)(id & 4095);
    int c = (int)((id >> 12) & 511);
    int b = (int)(id >> 21);
    outs[((size_t)b * 2560 + 512 + c) * 4096 + p] = pn[b * 512 + c];
}

// ---------------------------------------------------------------------------
// Small r x r conv (ds path). Split-K over 16 channel slices with f32 atomics.
// ---------------------------------------------------------------------------
__global__ void dsconv_init_kernel(const float* __restrict__ bias, float* __restrict__ y,
                                   int cout)
{
    size_t id = (size_t)blockIdx.x * blockDim.x + threadIdx.x;
    if (id >= (size_t)2 * cout * 4096) return;
    y[id] = bias[(int)((id >> 12) % cout)];
}

__global__ void dsconv_part_kernel(const float* __restrict__ x, const float* __restrict__ w,
                                   float* __restrict__ y, int cout, int r)
{
    size_t id = (size_t)blockIdx.x * blockDim.x + threadIdx.x;
    if (id >= (size_t)2 * 16 * cout * 4096) return;
    int xw = (int)(id & 63), yh = (int)((id >> 6) & 63);
    int rest = (int)(id >> 12);
    int o = rest % cout;
    int rest2 = rest / cout;
    int sl = rest2 & 15, b = rest2 >> 4;
    int pad = r >> 1, c0 = sl * 32;
    float acc = 0.f;
    for (int i2 = 0; i2 < r; ++i2) {
        int yi = yh + i2 - (pad - 1);
        if (yi < 0 || yi >= 64) continue;
        for (int j2 = 0; j2 < r; ++j2) {
            int xj = xw + j2 - (pad - 1);
            if (xj < 0 || xj >= 64) continue;
            const float* xp = x + ((size_t)b * 512 + c0) * 4096 + yi * 64 + xj;
            const float* wp = w + (((size_t)o * 512 + c0) * r + i2) * r + j2;
            for (int c = 0; c < 32; ++c)
                acc += xp[(size_t)c * 4096] * wp[(size_t)c * r * r];
        }
    }
    unsafeAtomicAdd(&y[(((size_t)b * cout + o) << 12) + yh * 64 + xw], acc);
}

// wrap-extension index map for the ds context
__device__ inline int wrapidx(int e, int lo, int hi)
{
    int pre = hi - lo;
    if (e < pre) return lo + e;
    e -= pre;
    if (e < 64) return e;
    return 64 - hi + (e - 64);
}

// kv[feat][gtok]; feat = (rh*r + rw)*cc + c
__global__ void kv_build_kernel(const float* __restrict__ ds, float* __restrict__ kv,
                                int r, int cc)
{
    size_t id = (size_t)blockIdx.x * blockDim.x + threadIdx.x;
    if (id >= (size_t)512 * 8192) return;
    int gtok = (int)(id & 8191);
    int feat = (int)(id >> 13);
    int c  = feat % cc;
    int rr = feat / cc;
    int rh = rr / r, rw = rr % r;
    int j = gtok & 7, i = (gtok >> 3) & 7, nw = (gtok >> 6) & 7, nh = (gtok >> 9) & 7, b = gtok >> 12;
    int lo = (r + 1) * 4, hi = r * 8;
    int hs = wrapidx(nh * 8 + i * r + rh, lo, hi);
    int ws = wrapidx(nw * 8 + j * r + rw, lo, hi);
    kv[id] = ds[(((size_t)b * cc + c) * 64 + hs) * 64 + ws];
}

// ---------------------------------------------------------------------------
// Fused windowed MHA, one block per (window, head). 16 heads, d = 32, L = 64.
// Q/K/V tiles staged via CDNA5 async global->LDS DMA (ASYNCcnt), converted to
// f16, then S = QK^T and O = PV computed with WMMA; f32 softmax in LDS.
// ---------------------------------------------------------------------------
__global__ __launch_bounds__(256)
void mha_kernel(const float* __restrict__ Qb, const float* __restrict__ Kb,
                const float* __restrict__ Vb, float* __restrict__ Ob, int Ntok)
{
    __shared__ float    stg[3 * 2048];  // raw f32 Q|K|V tiles (async landing zone)
    __shared__ _Float16 Qs[64][32];     // [token][d]
    __shared__ _Float16 Ks[64][32];     // [token][d]
    __shared__ _Float16 VsT[32][64];    // [d][token]
    __shared__ float    S[64][64];
    __shared__ _Float16 P[64][64];

    const int blk = blockIdx.x, head = blockIdx.y;
    const int t = threadIdx.x, lane = t & 31, wave = t >> 5;
    const int mloc = lane & 15, half = lane >> 4;

    // ---- async DMA: 1536 x 16B chunks, 6 per thread (uniform trip count)
    unsigned ldsBase = (unsigned)(unsigned long long)(void*)&stg[0];
    for (int ch = t; ch < 1536; ch += 256) {
        int arr = ch >> 9;            // 0:Q 1:K 2:V
        int rem = ch & 511;
        int dr  = rem >> 4;           // feature row 0..31
        int c4  = (rem & 15) * 4;     // token chunk
        const float* base = (arr == 0) ? Qb : ((arr == 1) ? Kb : Vb);
        const float* src = base + (size_t)(head * 32 + dr) * Ntok + (size_t)blk * 64 + c4;
        unsigned lds = ldsBase + (unsigned)((arr * 2048 + dr * 64 + c4) * 4);
        asm volatile("global_load_async_to_lds_b128 %0, %1, off"
                     :: "v"(lds), "v"(src) : "memory");
    }
    asm volatile("s_wait_asynccnt 0x0" ::: "memory");
    __syncthreads();

    // ---- convert f32 -> f16 operand tiles
    for (int e = t; e < 6144; e += 256) {
        int arr = e >> 11;
        int rem = e & 2047;
        int dr = rem >> 6, tk = rem & 63;
        _Float16 hv = (_Float16)stg[e];
        if (arr == 0)      Qs[tk][dr] = hv;
        else if (arr == 1) Ks[tk][dr] = hv;
        else               VsT[dr][tk] = hv;
    }
    __syncthreads();

    const float scale = 0.17677669529663687f;  // 1/sqrt(32)

    for (int tile = wave; tile < 16; tile += 8) {   // uniform per wave
        int mt = tile >> 2, nt = tile & 3;
        v16h af, bf; v8f c = {};
        #pragma unroll
        for (int v = 0; v < 8; ++v) {
            int ka = ((v & 4) ? 16 : 0) + half * 8 + (v & 3) * 2;
            af[2*v+0] = Qs[mt*16 + mloc][ka];  af[2*v+1] = Qs[mt*16 + mloc][ka+1];
            int kb = ((v & 4) ? 16 : 0) + half * 8 + (v & 3) * 2;
            bf[2*v+0] = Ks[nt*16 + mloc][kb];  bf[2*v+1] = Ks[nt*16 + mloc][kb+1];
        }
        c = WMMA_F16(af, bf, c);
        #pragma unroll
        for (int v = 0; v < 8; ++v)
            S[mt*16 + v + 8*half][nt*16 + mloc] = c[v] * scale;
    }
    __syncthreads();

    if (t < 64) {
        float mx = -1e30f;
        for (int k = 0; k < 64; ++k) mx = fmaxf(mx, S[t][k]);
        float sum = 0.f;
        for (int k = 0; k < 64; ++k) { float e = expf(S[t][k] - mx); S[t][k] = e; sum += e; }
        float rs = 1.f / sum;
        for (int k = 0; k < 64; ++k) P[t][k] = (_Float16)(S[t][k] * rs);
    }
    __syncthreads();

    {
        int mt = wave >> 1, nt = wave & 1;   // M: 4 tiles, N: 2 tiles (d = 32)
        v8f c = {};
        for (int ks = 0; ks < 64; ks += 32) {
            v16h af, bf;
            #pragma unroll
            for (int v = 0; v < 8; ++v) {
                int ka = ((v & 4) ? 16 : 0) + half * 8 + (v & 3) * 2;
                af[2*v+0] = P[mt*16 + mloc][ks + ka];      af[2*v+1] = P[mt*16 + mloc][ks + ka + 1];
                int kb = ((v & 4) ? 16 : 0) + half * 8 + (v & 3) * 2;
                bf[2*v+0] = VsT[nt*16 + mloc][ks + kb];    bf[2*v+1] = VsT[nt*16 + mloc][ks + kb + 1];
            }
            c = WMMA_F16(af, bf, c);
        }
        #pragma unroll
        for (int v = 0; v < 8; ++v) {
            int qt = mt * 16 + v + 8 * half;
            int d  = nt * 16 + mloc;
            Ob[(size_t)(head * 32 + d) * Ntok + (size_t)blk * 64 + qt] = c[v];
        }
    }
}

// Scatter attention output [c][gtok] -> outs[b][slot+c][h][w]
__global__ void scatter_attn_kernel(const float* __restrict__ pr, float* __restrict__ outs,
                                    int slotOff)
{
    size_t id = (size_t)blockIdx.x * blockDim.x + threadIdx.x;
    if (id >= (size_t)2 * 512 * 4096) return;
    int w = (int)(id & 63), h = (int)((id >> 6) & 63);
    int c = (int)((id >> 12) & 511), b = (int)(id >> 21);
    int gtok = (((b * 8 + (h >> 3)) * 8 + (w >> 3)) << 6) | ((h & 7) * 8 + (w & 7));
    outs[((size_t)b * 2560 + slotOff + c) * 4096 + h * 64 + w] = pr[(size_t)c * 8192 + gtok];
}

// Half-pixel bilinear 64x64 -> 128x128, writes into lowin channels [0,512)
__global__ void resize_kernel(const float* __restrict__ x, float* __restrict__ y)
{
    size_t id = (size_t)blockIdx.x * blockDim.x + threadIdx.x;
    if (id >= (size_t)2 * 512 * 16384) return;
    int xw = (int)(id & 127), yh = (int)((id >> 7) & 127);
    int c = (int)((id >> 14) & 511), b = (int)(id >> 23);
    float sy = yh * 0.5f - 0.25f;
    float sx = xw * 0.5f - 0.25f;
    int iy0 = (int)floorf(sy), ix0 = (int)floorf(sx);
    float fy = sy - iy0, fx = sx - ix0;
    int iy1 = min(max(iy0 + 1, 0), 63), ix1 = min(max(ix0 + 1, 0), 63);
    iy0 = min(max(iy0, 0), 63); ix0 = min(max(ix0, 0), 63);
    const float* xb = x + ((size_t)b * 512 + c) * 4096;
    float p00 = xb[iy0 * 64 + ix0], p01 = xb[iy0 * 64 + ix1];
    float p10 = xb[iy1 * 64 + ix0], p11 = xb[iy1 * 64 + ix1];
    float v = (1.f - fy) * ((1.f - fx) * p00 + fx * p01) + fy * ((1.f - fx) * p10 + fx * p11);
    y[((size_t)b * 560 + c) * 16384 + yh * 128 + xw] = v;
}

// ---------------------------------------------------------------------------
// Workspace layout (floats)
// ---------------------------------------------------------------------------
static constexpr size_t OFF_H4     = 0;                          // 2*512*256
static constexpr size_t OFF_H3     = OFF_H4     + 262144;        // 2*512*1024
static constexpr size_t OFF_H2     = OFF_H3     + 1048576;       // 2*512*4096
static constexpr size_t OFF_FIN    = OFF_H2     + 4194304;       // 2*1536*4096
static constexpr size_t OFF_C      = OFF_FIN    + 12582912;      // 2*512*4096
static constexpr size_t OFF_QBUF   = OFF_C      + 4194304;       // 512*8192
static constexpr size_t OFF_OUTS   = OFF_QBUF   + 4194304;       // 2*2560*4096
static constexpr size_t OFF_POOLED = OFF_OUTS   + 20971520;      // 1024
static constexpr size_t OFF_POOL2  = OFF_POOLED + 1024;          // 1024
static constexpr size_t OFF_POOLN  = OFF_POOL2  + 1024;          // 1024
static constexpr size_t OFF_DS     = OFF_POOLN  + 1024;          // 2*128*4096 max
static constexpr size_t OFF_KV     = OFF_DS     + 1048576;       // 512*8192
static constexpr size_t OFF_QP     = OFF_KV     + 4194304;
static constexpr size_t OFF_KP     = OFF_QP     + 4194304;
static constexpr size_t OFF_VP     = OFF_KP     + 4194304;
static constexpr size_t OFF_OP     = OFF_VP     + 4194304;
static constexpr size_t OFF_PR     = OFF_OP     + 4194304;
static constexpr size_t OFF_CAT    = OFF_PR     + 4194304;       // 2*512*4096
static constexpr size_t OFF_LOWIN  = OFF_CAT    + 4194304;       // 2*560*16384
static constexpr size_t OFF_TLOW   = OFF_LOWIN  + 18350080;      // 2*256*16384
static constexpr size_t OFF_TPROJ  = OFF_TLOW   + 8388608;       // 2*256*4096 max
static constexpr size_t OFF_GNS    = OFF_TPROJ  + 2097152;       // 64*8*2
static constexpr size_t OFF_GNMI   = OFF_GNS    + 1024;          // 64*2

extern "C" void kernel_launch(void* const* d_in, const int* in_sizes, int n_in,
                              void* d_out, int out_size, void* d_ws, size_t ws_size,
                              hipStream_t stream)
{
    (void)in_sizes; (void)n_in; (void)out_size; (void)ws_size;
    const float* c1 = (const float*)d_in[0];
    const float* c2 = (const float*)d_in[1];
    const float* c3 = (const float*)d_in[2];
    const float* c4 = (const float*)d_in[3];
    #define PF(i) ((const float*)d_in[i])
    const float *c4_w = PF(4),  *c4_g = PF(5),  *c4_b = PF(6);
    const float *c3_w = PF(7),  *c3_g = PF(8),  *c3_b = PF(9);
    const float *c2_w = PF(10), *c2_g = PF(11), *c2_b = PF(12);
    const float *fuse_w = PF(13), *fuse_g = PF(14), *fuse_b = PF(15);
    const float *short_w = PF(16), *short_g = PF(17), *short_b = PF(18);
    const float *pool_w = PF(19), *pool_g = PF(20), *pool_b = PF(21);
    const float *cat_w = PF(22), *cat_g = PF(23), *cat_b = PF(24);
    const float *low_w = PF(25), *low_g = PF(26), *low_b = PF(27);
    const float *c1_w = PF(28), *c1g = PF(29), *c1b = PF(30), *c1m = PF(31), *c1v = PF(32);
    const float *p2_w = PF(33), *p2_g = PF(34), *p2_b = PF(35);
    const float *p3_w = PF(36), *p3_g = PF(37), *p3_b = PF(38);
    const float *p4_w = PF(39), *p4_g = PF(40), *p4_b = PF(41);

    float* ws  = (float*)d_ws;
    float* out = (float*)d_out;
    float *H4 = ws + OFF_H4, *H3 = ws + OFF_H3, *H2 = ws + OFF_H2;
    float *FIN = ws + OFF_FIN, *CB = ws + OFF_C, *QBUF = ws + OFF_QBUF;
    float *OUTS = ws + OFF_OUTS, *POOLED = ws + OFF_POOLED, *POOL2 = ws + OFF_POOL2;
    float *POOLN = ws + OFF_POOLN, *DSB = ws + OFF_DS, *KV = ws + OFF_KV;
    float *QP = ws + OFF_QP, *KP = ws + OFF_KP, *VP = ws + OFF_VP;
    float *OP = ws + OFF_OP, *PR = ws + OFF_PR, *CAT = ws + OFF_CAT;
    float *LOWIN = ws + OFF_LOWIN, *TLOW = ws + OFF_TLOW, *TPROJ = ws + OFF_TPROJ;
    float *GNS = ws + OFF_GNS, *GNMI = ws + OFF_GNMI;

    auto NBLK = [](size_t total) { return (unsigned)((total + 255) / 256); };
    auto GEMM = [&](const float* A, const float* B, const float* bias, float* C,
                    int M, int K, int N, long long bBS, long long cBS, int batches) {
        dim3 g((N + 127) / 128, (M + 63) / 64, batches);
        gemm_wmma_kernel<<<g, 256, 0, stream>>>(A, B, bias, C, M, K, N, bBS, cBS);
    };
    auto GN = [&](const float* x, float* y, const float* g, const float* b,
                  int C, int HW, long long xBS, long long yBS) {
        gn_stat_kernel<<<dim3(64, 8), 256, 0, stream>>>(x, GNS, C, HW, xBS);
        gn_fin_kernel<<<1, 64, 0, stream>>>(GNS, GNMI, C, HW);
        gn_apply_kernel<<<NBLK((size_t)2 * C * HW), 256, 0, stream>>>(x, y, g, b, GNMI, C, HW, xBS, yBS);
    };

    // ---- stage lateral convs + GN + upsample/concat -> fuse
    GEMM(c4_w, c4, nullptr, H4, 512, 2048, 256, 2048LL * 256, 512LL * 256, 2);
    GN(H4, H4, c4_g, c4_b, 512, 256, 512LL * 256, 512LL * 256);
    GEMM(c3_w, c3, nullptr, H3, 512, 1024, 1024, 1024LL * 1024, 512LL * 1024, 2);
    GN(H3, H3, c3_g, c3_b, 512, 1024, 512LL * 1024, 512LL * 1024);
    GEMM(c2_w, c2, nullptr, H2, 512, 512, 4096, 512LL * 4096, 512LL * 4096, 2);
    GN(H2, H2, c2_g, c2_b, 512, 4096, 512LL * 4096, 512LL * 4096);
    fin_build_kernel<<<NBLK((size_t)2 * 1536 * 4096), 256, 0, stream>>>(H4, H3, H2, FIN);
    GEMM(fuse_w, FIN, nullptr, CB, 512, 1536, 4096, 1536LL * 4096, 512LL * 4096, 2);
    GN(CB, CB, fuse_g, fuse_b, 512, 4096, 512LL * 4096, 512LL * 4096);
    qbuf_build_kernel<<<NBLK((size_t)2 * 512 * 4096), 256, 0, stream>>>(CB, QBUF);

    // ---- outs slot 0: short path
    GEMM(short_w, CB, nullptr, OUTS, 512, 512, 4096, 512LL * 4096, 2560LL * 4096, 2);
    GN(OUTS, OUTS, short_g, short_b, 512, 4096, 2560LL * 4096, 2560LL * 4096);

    // ---- outs slot 1: global pool path
    pool_mean_kernel<<<1024, 256, 0, stream>>>(CB, POOLED);
    pool_fc_kernel<<<4, 256, 0, stream>>>(pool_w, POOLED, POOL2);
    GN(POOL2, POOLN, pool_g, pool_b, 512, 1, 512, 512);
    pool_broadcast_kernel<<<NBLK((size_t)2 * 512 * 4096), 256, 0, stream>>>(POOLN, OUTS);

    // ---- outs slots 2..4: windowed attention at r = 2, 4, 8
    const int rs[3] = {2, 4, 8};
    for (int ri = 0; ri < 3; ++ri) {
        int r = rs[ri], cc = 512 / (r * r);
        int base = 42 + ri * 10;
        const float *dsw = PF(base), *dsb = PF(base + 1);
        const float *bng = PF(base + 2), *bnb = PF(base + 3), *bnm = PF(base + 4), *bnv = PF(base + 5);
        const float *inw = PF(base + 6), *inb = PF(base + 7);
        const float *ow  = PF(base + 8), *ob  = PF(base + 9);

        dsconv_init_kernel<<<NBLK((size_t)2 * cc * 4096), 256, 0, stream>>>(dsb, DSB, cc);
        dsconv_part_kernel<<<NBLK((size_t)2 * 16 * cc * 4096), 256, 0, stream>>>(CB, dsw, DSB, cc, r);
        bn_relu_kernel<<<dim3(NBLK((size_t)cc * 4096), 1, 2), 256, 0, stream>>>(
            DSB, DSB, bng, bnb, bnm, bnv, cc, 4096, (long long)cc * 4096, (long long)cc * 4096);
        kv_build_kernel<<<NBLK((size_t)512 * 8192), 256, 0, stream>>>(DSB, KV, r, cc);

        GEMM(inw,                 QBUF, inb,        QP, 512, 512, 8192, 0, 0, 1);
        GEMM(inw + 512 * 512,     KV,   inb + 512,  KP, 512, 512, 8192, 0, 0, 1);
        GEMM(inw + 2 * 512 * 512, KV,   inb + 1024, VP, 512, 512, 8192, 0, 0, 1);
        mha_kernel<<<dim3(128, 16), 256, 0, stream>>>(QP, KP, VP, OP, 8192);
        GEMM(ow, OP, ob, PR, 512, 512, 8192, 0, 0, 1);
        scatter_attn_kernel<<<NBLK((size_t)2 * 512 * 4096), 256, 0, stream>>>(
            PR, OUTS, 1024 + 512 * ri);
    }

    // ---- cat conv + GN, bilinear up to 128x128
    GEMM(cat_w, OUTS, nullptr, CAT, 512, 2560, 4096, 2560LL * 4096, 512LL * 4096, 2);
    GN(CAT, CAT, cat_g, cat_b, 512, 4096, 512LL * 4096, 512LL * 4096);
    resize_kernel<<<NBLK((size_t)2 * 512 * 16384), 256, 0, stream>>>(CAT, LOWIN);

    // ---- c1 branch -> lowin channels [512, 560)
    GEMM(c1_w, c1, nullptr, TPROJ, 48, 256, 16384, 256LL * 16384, 48LL * 16384, 2);
    bn_relu_kernel<<<dim3(NBLK((size_t)48 * 16384), 1, 2), 256, 0, stream>>>(
        TPROJ, LOWIN + (size_t)512 * 16384, c1g, c1b, c1m, c1v,
        48, 16384, 48LL * 16384, 560LL * 16384);

    // ---- low conv + GN -> output 0
    GEMM(low_w, LOWIN, nullptr, TLOW, 256, 560, 16384, 560LL * 16384, 256LL * 16384, 2);
    GN(TLOW, out, low_g, low_b, 256, 16384, 256LL * 16384, 256LL * 16384);

    // ---- projections -> outputs 1..3 (proj4, proj3, proj2)
    GEMM(p4_w, c4, nullptr, TPROJ, 256, 2048, 256, 2048LL * 256, 256LL * 256, 2);
    GN(TPROJ, out + 8388608, p4_g, p4_b, 256, 256, 256LL * 256, 256LL * 256);
    GEMM(p3_w, c3, nullptr, TPROJ, 256, 1024, 1024, 1024LL * 1024, 256LL * 1024, 2);
    GN(TPROJ, out + 8519680, p3_g, p3_b, 256, 1024, 256LL * 1024, 256LL * 1024);
    GEMM(p2_w, c2, nullptr, TPROJ, 256, 512, 4096, 512LL * 4096, 256LL * 4096, 2);
    GN(TPROJ, out + 9043968, p2_g, p2_b, 256, 4096, 256LL * 4096, 256LL * 4096);
    #undef PF
}
